// Model_55138790146400
// MI455X (gfx1250) — matrix operand order repaired and checked by reading, not compiled, not run on hardware
//
#include <hip/hip_runtime.h>

#define NN 16500
#define NE 100000
#define KT3 3

typedef __attribute__((ext_vector_type(16))) _Float16 v16h;
typedef __attribute__((ext_vector_type(8)))  _Float16 v8h;
typedef __attribute__((ext_vector_type(4)))  _Float16 v4h;
typedef __attribute__((ext_vector_type(8)))  float    v8f;
typedef __attribute__((ext_vector_type(4)))  float    v4f;

// ---------------------------------------------------------------------------
// Edge softmax scatter: one wave per edge, lanes stride over channels.
// msg = relu(x[src] + ea*We + be) + eps ; den += e^msg, num += msg*e^msg
// (max-subtraction elided: msg bounded, exp cannot overflow; ratio identical)
// ---------------------------------------------------------------------------
__global__ __launch_bounds__(256) void edge_scatter(
    const float* __restrict__ x, int lda, int ci,
    const int* __restrict__ src, const int* __restrict__ dst,
    const float* __restrict__ ea,
    const float* __restrict__ We, const float* __restrict__ be,
    float* __restrict__ num, float* __restrict__ den)
{
    const int wave = threadIdx.x >> 5;
    const int lane = threadIdx.x & 31;
    const int e = blockIdx.x * 8 + wave;
    const int total = NE + NN;
    if (e >= total) return;
    int s, d; float a;
    if (e < NE) { s = src[e]; d = dst[e]; a = ea[e]; }
    else        { s = e - NE; d = s;      a = 1.0f;  }
    const float* xr = x + (size_t)s * lda;
    float* nr = num + (size_t)d * 256;
    float* dr = den + (size_t)d * 256;
    for (int c = lane; c < ci; c += 32) {
        float msg = fmaxf(xr[c] + a * We[c] + be[c], 0.0f) + 1e-7f;
        float w = __expf(msg);
        atomicAdd(&dr[c], w);
        atomicAdd(&nr[c], msg * w);
    }
}

// out = num/den + x   (in place over num; den>0 guaranteed by self-loop)
__global__ __launch_bounds__(256) void finalize_out(
    float* __restrict__ num, const float* __restrict__ den,
    const float* __restrict__ x, int lda, int ci)
{
    const int row = blockIdx.x;
    for (int c = threadIdx.x; c < ci; c += blockDim.x) {
        size_t o = (size_t)row * 256 + c;
        num[o] = num[o] / den[o] + x[(size_t)row * lda + c];
    }
}

// Per-channel train-mode BN stats over N rows; fold gamma/beta into scale/shift.
__global__ __launch_bounds__(256) void bn_stats(
    const _Float16* __restrict__ h, int ldh,
    const float* __restrict__ g, const float* __restrict__ bt,
    float* __restrict__ scale, float* __restrict__ shift)
{
    const int c = blockIdx.x;
    float s = 0.f, sq = 0.f;
    for (int r = threadIdx.x; r < NN; r += blockDim.x) {
        float v = (float)h[(size_t)r * ldh + c];
        s += v; sq += v * v;
    }
    __shared__ float ss[256], qq[256];
    ss[threadIdx.x] = s; qq[threadIdx.x] = sq;
    __syncthreads();
    for (int st = 128; st > 0; st >>= 1) {
        if (threadIdx.x < st) { ss[threadIdx.x] += ss[threadIdx.x + st];
                                qq[threadIdx.x] += qq[threadIdx.x + st]; }
        __syncthreads();
    }
    if (threadIdx.x == 0) {
        float mu  = ss[0] * (1.0f / NN);
        float var = qq[0] * (1.0f / NN) - mu * mu;
        float sc  = g[c] * rsqrtf(var + 1e-5f);
        scale[c] = sc;
        shift[c] = bt[c] - mu * sc;
    }
}

__global__ __launch_bounds__(256) void leaky_inplace(float* __restrict__ x, int n)
{
    int i = blockIdx.x * blockDim.x + threadIdx.x;
    if (i < n) { float v = x[i]; x[i] = v > 0.f ? v : 0.01f * v; }
}

// ---------------------------------------------------------------------------
// WMMA GEMM: C[M,Ncols] (+)= act(A[M,Kd]) @ B[Kd,Ncols] + bias
// AMODE 0: A fp32 plain; AMODE 1: A f16 with fused BN(scale,shift)+ReLU
// OUTF16 : store f16 ; ACC: accumulate into fp32 C
// Block: 256 threads = 8 waves; tile 128x64; wave w owns rows [16w,16w+16) x 64 cols.
// All LDS fragment traffic is contiguous 16B (ds_load/store_b128):
//  - A fragment = two v8h runs per the ISA 16-bit A layout
//  - B stored transposed (Bst[n][k]) so a column's K-run is contiguous
// ---------------------------------------------------------------------------
#define MT 128
#define NT 64
#define KC 32
#define LDK (KC + 8)   // 40 halves = 80B row stride (16B aligned)

template<int AMODE, int OUTF16, int ACC>
__global__ __launch_bounds__(256) void gemm_wmma(
    const void* __restrict__ A_, int lda,
    const float* __restrict__ B, int ldb,
    void* __restrict__ C_, int ldc,
    const float* __restrict__ bias,
    const float* __restrict__ scale, const float* __restrict__ shift,
    int M, int Kd, int Ncols)
{
    __shared__ __align__(16) _Float16 As[MT][LDK];
    __shared__ __align__(16) _Float16 Bst[NT][LDK];   // transposed: [n][k]
    const int tid  = threadIdx.x;
    const int wave = tid >> 5, lane = tid & 31;
    const int m0 = blockIdx.x * MT;
    const int n0 = blockIdx.y * NT;
    const int r  = lane & 15;
    const int hi = lane >> 4;           // 0: lanes 0-15, 1: lanes 16-31

    v8f acc0 = {}, acc1 = {}, acc2 = {}, acc3 = {};

    const int nK = (Kd + KC - 1) / KC;
    for (int kc = 0; kc < nK; ++kc) {
        const int kb = kc * KC;

        // ---- stage A tile (128x32): 512 groups of 8 halves, 2/thread ----
        for (int gidx = tid; gidx < MT * (KC / 8); gidx += 256) {
            const int rr = gidx >> 2, kk8 = (gidx & 3) * 8;
            const int gr = m0 + rr, gk = kb + kk8;
            v8h hv;
            if (gr < M && gk + 7 < Kd) {
                if (AMODE == 0) {
                    const float* ap = (const float*)A_ + (size_t)gr * lda + gk;
                    v4f f0 = *(const v4f*)(ap);
                    v4f f1 = *(const v4f*)(ap + 4);
                    #pragma unroll
                    for (int j = 0; j < 4; ++j) {
                        hv[j]     = (_Float16)f0[j];
                        hv[4 + j] = (_Float16)f1[j];
                    }
                } else {
                    const _Float16* ap = (const _Float16*)A_ + (size_t)gr * lda + gk;
                    v4h h0 = *(const v4h*)(ap);       // 8B aligned (ldh may be 340)
                    v4h h1 = *(const v4h*)(ap + 4);
                    #pragma unroll
                    for (int j = 0; j < 4; ++j) {
                        hv[j]     = (_Float16)fmaxf((float)h0[j] * scale[gk + j]     + shift[gk + j],     0.f);
                        hv[4 + j] = (_Float16)fmaxf((float)h1[j] * scale[gk + 4 + j] + shift[gk + 4 + j], 0.f);
                    }
                }
            } else {
                #pragma unroll
                for (int j = 0; j < 8; ++j) {
                    const int gkj = gk + j;
                    float v = 0.f;
                    if (gr < M && gkj < Kd) {
                        if (AMODE == 0) {
                            v = ((const float*)A_)[(size_t)gr * lda + gkj];
                        } else {
                            float t = (float)((const _Float16*)A_)[(size_t)gr * lda + gkj];
                            v = fmaxf(t * scale[gkj] + shift[gkj], 0.f);
                        }
                    }
                    hv[j] = (_Float16)v;
                }
            }
            *(v8h*)&As[rr][kk8] = hv;   // ds_store_b128
        }

        // ---- stage B tile transposed (64 cols x 32 k): 256 groups, 1/thread ----
        {
            const int nn = tid & (NT - 1), kk8 = (tid >> 6) * 8;
            const int gn = n0 + nn, gk = kb + kk8;
            v8h hv;
            #pragma unroll
            for (int j = 0; j < 8; ++j) {
                float v = (gk + j < Kd && gn < Ncols) ? B[(size_t)(gk + j) * ldb + gn] : 0.f;
                hv[j] = (_Float16)v;
            }
            *(v8h*)&Bst[nn][kk8] = hv;  // ds_store_b128
        }
        __syncthreads();

        // ---- fragments: contiguous b128 LDS loads ----
        const int ar = wave * 16 + r;
        const int kbase = hi * 8;       // low half-wave: K 0-7/16-23; high: 8-15/24-31
        v8h a0 = *(const v8h*)&As[ar][kbase];
        v8h a1 = *(const v8h*)&As[ar][kbase + 16];
        v16h af = __builtin_shufflevector(a0, a1, 0,1,2,3,4,5,6,7,8,9,10,11,12,13,14,15);

        const int krow = hi * 8;        // elem j<8 = B[K=krow+j][col], j>=8 = B[K=krow+8+j][col]
        v8h b0lo = *(const v8h*)&Bst[r][krow];
        v8h b0hi = *(const v8h*)&Bst[r][krow + 16];
        v8h b1lo = *(const v8h*)&Bst[16 + r][krow];
        v8h b1hi = *(const v8h*)&Bst[16 + r][krow + 16];
        v8h b2lo = *(const v8h*)&Bst[32 + r][krow];
        v8h b2hi = *(const v8h*)&Bst[32 + r][krow + 16];
        v8h b3lo = *(const v8h*)&Bst[48 + r][krow];
        v8h b3hi = *(const v8h*)&Bst[48 + r][krow + 16];
        v16h bf0 = __builtin_shufflevector(b0lo, b0hi, 0,1,2,3,4,5,6,7,8,9,10,11,12,13,14,15);
        v16h bf1 = __builtin_shufflevector(b1lo, b1hi, 0,1,2,3,4,5,6,7,8,9,10,11,12,13,14,15);
        v16h bf2 = __builtin_shufflevector(b2lo, b2hi, 0,1,2,3,4,5,6,7,8,9,10,11,12,13,14,15);
        v16h bf3 = __builtin_shufflevector(b3lo, b3hi, 0,1,2,3,4,5,6,7,8,9,10,11,12,13,14,15);

        acc0 = __builtin_amdgcn_wmma_f32_16x16x32_f16(false, af, false, bf0, (short)0, acc0, false, false);
        acc1 = __builtin_amdgcn_wmma_f32_16x16x32_f16(false, af, false, bf1, (short)0, acc1, false, false);
        acc2 = __builtin_amdgcn_wmma_f32_16x16x32_f16(false, af, false, bf2, (short)0, acc2, false, false);
        acc3 = __builtin_amdgcn_wmma_f32_16x16x32_f16(false, af, false, bf3, (short)0, acc3, false, false);
        __syncthreads();
    }

    // Epilogue: C/D layout — VGPR v: row = v (+8 for high half-wave), col = lane%16
    const int rowbase = m0 + wave * 16 + hi * 8;
    const int colbase = n0 + r;
    #pragma unroll
    for (int v = 0; v < 8; ++v) {
        const int row = rowbase + v;
        if (row < M) {
            float vals[4] = {acc0[v], acc1[v], acc2[v], acc3[v]};
            #pragma unroll
            for (int t = 0; t < 4; ++t) {
                const int col = colbase + t * 16;
                if (col < Ncols) {
                    float o = vals[t] + bias[col];
                    if (OUTF16) {
                        ((_Float16*)C_)[(size_t)row * ldc + col] = (_Float16)o;
                    } else {
                        float* Cp = (float*)C_;
                        size_t off = (size_t)row * ldc + col;
                        if (ACC) Cp[off] += o; else Cp[off] = o;
                    }
                }
            }
        }
    }
}

// ---------------------------------------------------------------------------
extern "C" void kernel_launch(void* const* d_in, const int* in_sizes, int n_in,
                              void* d_out, int out_size, void* d_ws, size_t ws_size,
                              hipStream_t stream)
{
    (void)in_sizes; (void)n_in; (void)out_size; (void)ws_size;
    const float* x_in  = (const float*)d_in[0];
    const int*   eidx  = (const int*)d_in[1];     // [K,2,E]
    const float* eattr = (const float*)d_in[2];   // [K,E,1]

    // workspace layout (fp32 unless noted): 4x [NN,256] + f16 [NN,512] + stats
    float* ws = (float*)d_ws;
    const size_t NB = (size_t)NN * 256;
    float* xbuf0 = ws;
    float* xbuf1 = ws + NB;
    float* numb  = ws + 2 * NB;   // becomes "out" after finalize
    float* denb  = ws + 3 * NB;
    _Float16* hb = (_Float16*)(ws + 4 * NB);            // [NN, 512] f16
    float* scaleb = (float*)(hb + (size_t)NN * 512);
    float* shiftb = scaleb + 512;

    const int ci_arr[3] = {170, 256, 256};
    const float* curx = x_in;
    int lda = 170;

    for (int l = 0; l < 3; ++l) {
        const int ci = ci_arr[l], c2 = 2 * ci, co = 256;
        const float* We = (const float*)d_in[3 + 8 * l + 0];
        const float* be = (const float*)d_in[3 + 8 * l + 1];
        const float* Wa = (const float*)d_in[3 + 8 * l + 2];
        const float* ba = (const float*)d_in[3 + 8 * l + 3];
        const float* g  = (const float*)d_in[3 + 8 * l + 4];
        const float* bt = (const float*)d_in[3 + 8 * l + 5];
        const float* Wb = (const float*)d_in[3 + 8 * l + 6];
        const float* bb = (const float*)d_in[3 + 8 * l + 7];

        float* ybuf = (l == 0) ? xbuf0 : (l == 1) ? xbuf1 : (float*)d_out;

        for (int k = 0; k < KT3; ++k) {
            hipMemsetAsync(numb, 0, NB * sizeof(float), stream);
            hipMemsetAsync(denb, 0, NB * sizeof(float), stream);

            const int* src = eidx + (size_t)k * 2 * NE;
            const int* dst = src + NE;
            const float* ea = eattr + (size_t)k * NE;
            const int totE = NE + NN;
            edge_scatter<<<(totE + 7) / 8, 256, 0, stream>>>(
                curx, lda, ci, src, dst, ea, We + (size_t)k * ci, be + (size_t)k * ci,
                numb, denb);
            finalize_out<<<NN, 256, 0, stream>>>(numb, denb, curx, lda, ci);

            // h = out @ Wa + ba  -> f16 [NN, c2]
            dim3 g1((NN + MT - 1) / MT, (c2 + NT - 1) / NT);
            gemm_wmma<0, 1, 0><<<g1, 256, 0, stream>>>(
                numb, 256, Wa + (size_t)k * ci * c2, c2,
                hb, c2, ba + (size_t)k * c2, nullptr, nullptr, NN, ci, c2);

            bn_stats<<<c2, 256, 0, stream>>>(hb, c2, g + (size_t)k * c2,
                                             bt + (size_t)k * c2, scaleb, shiftb);

            // y (+)= relu(bn(h)) @ Wb + bb   (k==0 overwrites, else accumulates)
            dim3 g2((NN + MT - 1) / MT, (co + NT - 1) / NT);
            if (k == 0)
                gemm_wmma<1, 0, 0><<<g2, 256, 0, stream>>>(
                    hb, c2, Wb + (size_t)k * c2 * co, co,
                    ybuf, 256, bb + (size_t)k * co, scaleb, shiftb, NN, c2, co);
            else
                gemm_wmma<1, 0, 1><<<g2, 256, 0, stream>>>(
                    hb, c2, Wb + (size_t)k * c2 * co, co,
                    ybuf, 256, bb + (size_t)k * co, scaleb, shiftb, NN, c2, co);
        }

        if (l < 2) {
            leaky_inplace<<<((int)NB + 255) / 256, 256, 0, stream>>>(ybuf, (int)NB);
            curx = ybuf; lda = 256;
        }
    }
}
